// SeaIceAdr_x5_5k3_t3_w128_d3_52570399703504
// MI455X (gfx1250) — matrix operand order repaired and checked by reading, not compiled, not run on hardware
//
#include <hip/hip_runtime.h>
#include <hip/hip_bf16.h>

// ---------------------------------------------------------------------------
// Types for WMMA operands (wave32, CDNA5)
// ---------------------------------------------------------------------------
typedef _Float16 v16h __attribute__((ext_vector_type(16)));
typedef _Float16 v8h  __attribute__((ext_vector_type(8)));
typedef float    v8f  __attribute__((ext_vector_type(8)));
typedef float    f4u  __attribute__((ext_vector_type(4), aligned(4)));  // dword-aligned vec load

// Hardware tanh (v_tanh_f32) when the toolchain exposes it; else libm expansion.
#if __has_builtin(__builtin_amdgcn_tanhf)
__device__ __forceinline__ float fast_tanh(float x) { return __builtin_amdgcn_tanhf(x); }
#else
__device__ __forceinline__ float fast_tanh(float x) { return tanhf(x); }
#endif

// Async global->LDS 16B copy (ASYNCcnt path, no VGPR round-trip).
__device__ __forceinline__ void async_copy_b128(unsigned lds_off, const void* gsrc) {
    asm volatile("global_load_async_to_lds_b128 %0, %1, off"
                 :: "v"(lds_off), "v"(gsrc) : "memory");
}
__device__ __forceinline__ void wait_async0() {
    asm volatile("s_wait_asynccnt 0" ::: "memory");
}
__device__ __forceinline__ unsigned lds_off_of(const void* p) {
    // Generic pointer to LDS: addr[31:0] is the LDS byte offset (ISA aperture rule).
    return (unsigned)(size_t)p;
}

// Workspace layout (all _Float16):
//   [0      , 49152)  W_eff  : 128 x 384 (K padded 363->384, row-major [o][k])
//   [49152  , 65536)  w1     : 128 x 128 (row-major [o][k])
//   [65536  , 81920)  w2
//   [81920  , 98304)  w3
//   [98304  ,100352)  wf     : 16  x 128 (rows 5..15 zero)
#define WS_WEFF   0
#define WS_W1     49152
#define WS_WF     98304
#define WS_TOTAL  100352

// ---------------------------------------------------------------------------
// Kernel 1: fold the 5 stencils into mix_a -> W_eff, convert weights to f16.
// ---------------------------------------------------------------------------
__global__ void prep_kernel(const float* __restrict__ mix_a,
                            const float* __restrict__ w1,
                            const float* __restrict__ w2,
                            const float* __restrict__ w3,
                            const float* __restrict__ wf,
                            _Float16* __restrict__ wh) {
    int idx = blockIdx.x * blockDim.x + threadIdx.x;
    if (idx >= WS_TOTAL) return;

    if (idx < WS_W1) {
        // W_eff[o, c,y,x] = sum_{s,i,j} mix_a[o, s*243 + c*81 + (y-i)*9 + (x-j)] * k_s[i,j]
        const int es[11] = {0, 1,1, 2,2, 3,3,3, 4,4,4};
        const int ei[11] = {1, 0,2, 1,1, 0,1,2, 1,1,1};
        const int ej[11] = {1, 1,1, 0,2, 1,1,1, 0,1,2};
        const float ev[11] = {1.f, -5.f, 5.f, -5.f, 5.f,
                              100.f, -200.f, 100.f, 100.f, -200.f, 100.f};
        int o = idx / 384, kp = idx % 384;
        float v = 0.f;
        if (kp < 363) {
            int c = kp / 121, rem = kp - c * 121;
            int y = rem / 11, x = rem - y * 11;
            const float* ma = mix_a + (size_t)o * 1215;
            #pragma unroll
            for (int e = 0; e < 11; ++e) {
                int n = y - ei[e], m = x - ej[e];
                if (n >= 0 && n < 9 && m >= 0 && m < 9)
                    v += ma[es[e] * 243 + c * 81 + n * 9 + m] * ev[e];
            }
        }
        wh[idx] = (_Float16)v;
    } else if (idx < WS_WF) {
        int e = idx - WS_W1;
        int which = e / 16384;  e -= which * 16384;
        const float* w = (which == 0) ? w1 : (which == 1) ? w2 : w3;
        wh[idx] = (_Float16)w[e];
    } else {
        int e = idx - WS_WF;
        int o = e / 128, k = e - o * 128;
        wh[idx] = (_Float16)((o < 5) ? wf[o * 128 + k] : 0.f);
    }
}

// ---------------------------------------------------------------------------
// Kernel 2: fused MLP. 256 threads = 8 waves; wave owns 16 batch rows.
// LDS (dynamic, 132KB): s_wb (49152 h) holds W_eff for layer0, then w1|w2|w3;
//                       s_wf (2048 h); s_x (8 waves * 16x128 activation slices)
// ---------------------------------------------------------------------------
#define L_WB 0
#define L_WF 49152
#define L_X  51200
#define L_TOTAL 67584  // halves -> 135168 bytes

__global__ __launch_bounds__(256, 1)
void fused_kernel(const float* __restrict__ data,
                  const float* __restrict__ mix_b,
                  const float* __restrict__ b1,
                  const float* __restrict__ b2,
                  const float* __restrict__ b3,
                  const float* __restrict__ bf,
                  const _Float16* __restrict__ wh,
                  float* __restrict__ out) {
    extern __shared__ __attribute__((aligned(128))) _Float16 smem[];
    _Float16* s_wb = smem + L_WB;
    _Float16* s_wf = smem + L_WF;
    _Float16* s_x  = smem + L_X;

    const int tid = threadIdx.x;

    // ---- Stage W_eff + wf into LDS via async global->LDS DMA ----
    {
        const uint4* src = (const uint4*)wh;
        const unsigned dwb = lds_off_of(s_wb);
        const unsigned dwf = lds_off_of(s_wf);
        for (int i = tid; i < 6144; i += 256)
            async_copy_b128(dwb + 16u * i, src + i);                    // W_eff
        async_copy_b128(dwf + 16u * tid, src + 12288 + tid);            // wf
        wait_async0();
    }
    __syncthreads();

    const int wave = tid >> 5;
    const int lane = tid & 31;
    const int lm   = lane & 15;   // row index (A/C) / col index (B/C)
    const int g    = lane >> 4;   // lane half-group
    const int rowBase = blockIdx.x * 128 + wave * 16;
    _Float16* xs = s_x + wave * (16 * 128);   // wave-private 16x128 slice

    v8f acc[8];

    // ================= Layer 0: mixed = data @ W_eff^T + mix_b ==============
    #pragma unroll
    for (int nt = 0; nt < 8; ++nt) {
        float bv = mix_b[nt * 16 + lm];
        #pragma unroll
        for (int r = 0; r < 8; ++r) acc[nt][r] = bv;
    }
    const float* drow = data + (size_t)(rowBase + lm) * 363;
    for (int kc = 0; kc < 12; ++kc) {
        // A fragment: lane holds row lm, halves = K chunks [16g..16g+7], [16g+8..16g+15]
        v16h a;
        const int c0 = kc * 32 + 16 * g;
        const int c1 = c0 + 8;
        if (kc < 11) {
            f4u u0 = *(const f4u*)(drow + c0);
            f4u u1 = *(const f4u*)(drow + c0 + 4);
            f4u u2 = *(const f4u*)(drow + c1);
            f4u u3 = *(const f4u*)(drow + c1 + 4);
            #pragma unroll
            for (int h = 0; h < 4; ++h) {
                a[h]      = (_Float16)u0[h];
                a[h + 4]  = (_Float16)u1[h];
                a[h + 8]  = (_Float16)u2[h];
                a[h + 12] = (_Float16)u3[h];
            }
        } else {  // tail chunk: guard against K >= 363
            #pragma unroll
            for (int h = 0; h < 8; ++h) {
                int k0 = c0 + h, k1 = c1 + h;
                a[h]     = (_Float16)((k0 < 363) ? drow[k0] : 0.f);
                a[h + 8] = (_Float16)((k1 < 363) ? drow[k1] : 0.f);
            }
        }
        #pragma unroll
        for (int nt = 0; nt < 8; ++nt) {
            const v16h bfr = *(const v16h*)(s_wb + (nt * 16 + lm) * 384 + kc * 32 + 16 * g);
            acc[nt] = __builtin_amdgcn_wmma_f32_16x16x32_f16(
                false, a, false, bfr, (short)0, acc[nt], false, false);
        }
    }
    // tanh -> activation slice (C/D layout: reg r -> row r+8g, col nt*16+lm)
    #pragma unroll
    for (int nt = 0; nt < 8; ++nt)
        #pragma unroll
        for (int r = 0; r < 8; ++r)
            xs[(r + 8 * g) * 128 + nt * 16 + lm] = (_Float16)fast_tanh(acc[nt][r]);

    // ---- Swap LDS weights: W_eff -> w1|w2|w3 (async DMA) ----
    __syncthreads();
    {
        const uint4* src = (const uint4*)wh;
        const unsigned dwb = lds_off_of(s_wb);
        for (int i = tid; i < 6144; i += 256)
            async_copy_b128(dwb + 16u * i, src + 6144 + i);
        wait_async0();
    }
    __syncthreads();

    // ================= Hidden layers: x = tanh(x @ wl^T + bl) ===============
    const float* hb[3] = {b1, b2, b3};
    for (int l = 0; l < 3; ++l) {
        #pragma unroll
        for (int nt = 0; nt < 8; ++nt) {
            float bv = hb[l][nt * 16 + lm];
            #pragma unroll
            for (int r = 0; r < 8; ++r) acc[nt][r] = bv;
        }
        const _Float16* wl = s_wb + l * 16384;
        #pragma unroll
        for (int kc = 0; kc < 4; ++kc) {
            v16h a;
            const v8h lo = *(const v8h*)(xs + lm * 128 + kc * 32 + 16 * g);
            const v8h hi = *(const v8h*)(xs + lm * 128 + kc * 32 + 8 + 16 * g);
            #pragma unroll
            for (int h = 0; h < 8; ++h) { a[h] = lo[h]; a[h + 8] = hi[h]; }
            #pragma unroll
            for (int nt = 0; nt < 8; ++nt) {
                const v16h bfr = *(const v16h*)(wl + (nt * 16 + lm) * 128 + kc * 32 + 16 * g);
                acc[nt] = __builtin_amdgcn_wmma_f32_16x16x32_f16(
                    false, a, false, bfr, (short)0, acc[nt], false, false);
            }
        }
        // in-place update is safe: wave-local slice, DS ops are in-order
        #pragma unroll
        for (int nt = 0; nt < 8; ++nt)
            #pragma unroll
            for (int r = 0; r < 8; ++r)
                xs[(r + 8 * g) * 128 + nt * 16 + lm] = (_Float16)fast_tanh(acc[nt][r]);
    }

    // ================= Final layer: out = x @ wf^T + bf (5 cols) ============
    v8f accf;
    {
        float bv = (lm < 5) ? bf[lm] : 0.f;
        #pragma unroll
        for (int r = 0; r < 8; ++r) accf[r] = bv;
    }
    #pragma unroll
    for (int kc = 0; kc < 4; ++kc) {
        v16h a;
        const v8h lo = *(const v8h*)(xs + lm * 128 + kc * 32 + 16 * g);
        const v8h hi = *(const v8h*)(xs + lm * 128 + kc * 32 + 8 + 16 * g);
        #pragma unroll
        for (int h = 0; h < 8; ++h) { a[h] = lo[h]; a[h + 8] = hi[h]; }
        const v16h bfr = *(const v16h*)(s_wf + lm * 128 + kc * 32 + 16 * g);
        accf = __builtin_amdgcn_wmma_f32_16x16x32_f16(
            false, a, false, bfr, (short)0, accf, false, false);
    }
    if (lm < 5) {
        #pragma unroll
        for (int r = 0; r < 8; ++r)
            out[(size_t)(rowBase + r + 8 * g) * 5 + lm] = accf[r];
    }
}

// ---------------------------------------------------------------------------
// Host entry
// ---------------------------------------------------------------------------
extern "C" void kernel_launch(void* const* d_in, const int* in_sizes, int n_in,
                              void* d_out, int out_size, void* d_ws, size_t ws_size,
                              hipStream_t stream) {
    const float* data  = (const float*)d_in[0];
    const float* mix_a = (const float*)d_in[1];
    const float* mix_b = (const float*)d_in[2];
    const float* w1    = (const float*)d_in[3];
    const float* b1    = (const float*)d_in[4];
    const float* w2    = (const float*)d_in[5];
    const float* b2    = (const float*)d_in[6];
    const float* w3    = (const float*)d_in[7];
    const float* b3    = (const float*)d_in[8];
    const float* wf    = (const float*)d_in[9];
    const float* bf    = (const float*)d_in[10];
    float* out = (float*)d_out;
    _Float16* wh = (_Float16*)d_ws;

    const int B = in_sizes[0] / 363;           // 65536

    prep_kernel<<<(WS_TOTAL + 255) / 256, 256, 0, stream>>>(mix_a, w1, w2, w3, wf, wh);

    const size_t shmem = (size_t)L_TOTAL * sizeof(_Float16);  // 132 KB
    (void)hipFuncSetAttribute(reinterpret_cast<const void*>(fused_kernel),
                              hipFuncAttributeMaxDynamicSharedMemorySize, (int)shmem);
    fused_kernel<<<B / 128, 256, shmem, stream>>>(data, mix_b, b1, b2, b3, bf, wh, out);
}
